// MoEBlock_7825430413738
// MI455X (gfx1250) — matrix operand order repaired and checked by reading, not compiled, not run on hardware
//
#include <hip/hip_runtime.h>
#include <hip/hip_bf16.h>
#include <math.h>

// ---- shapes -----------------------------------------------------------------
#define DD 1024      // model dim
#define FF 4096      // ffn dim
#define EE 8         // routed experts
#define TT 8192      // tokens (2*4096)
#define NB 9         // buckets: 8 routed + 1 shared
#define MT 32        // tokens per block tile (2 x 16-row WMMA tiles)

typedef __attribute__((ext_vector_type(16))) __bf16 v16bf;
typedef __attribute__((ext_vector_type(8)))  __bf16 v8bf;
typedef __attribute__((ext_vector_type(4)))  __bf16 v4bf;
typedef __attribute__((ext_vector_type(8)))  float  v8f;

// ---- workspace layout (bytes) ----------------------------------------------
#define OFF_W2   75497472ull
#define OFF_CNT  150994944ull
#define OFF_TOK  150995200ull
#define OFF_WL   151257344ull

// ---- zero fill --------------------------------------------------------------
__global__ void k_zero(float4* p, int n4) {
    int stride = gridDim.x * blockDim.x;
    float4 z; z.x = 0.f; z.y = 0.f; z.z = 0.f; z.w = 0.f;
    for (int i = blockIdx.x * blockDim.x + threadIdx.x; i < n4; i += stride)
        p[i] = z;
}

// ---- fp32 -> bf16 weight conversion (makes all weights L2-resident) ---------
__global__ void k_convert(const float* __restrict__ w1, const float* __restrict__ sw1,
                          const float* __restrict__ w2, const float* __restrict__ sw2,
                          unsigned short* __restrict__ w1bf, unsigned short* __restrict__ w2bf) {
    const size_t Q  = (size_t)NB * FF * DD / 4;
    const size_t E8 = (size_t)EE * FF * DD;
    size_t stride = (size_t)gridDim.x * blockDim.x;
    for (size_t i = (size_t)blockIdx.x * blockDim.x + threadIdx.x; i < 2 * Q; i += stride) {
        int region = (i >= Q);
        size_t j = (region ? i - Q : i) * 4;
        const float* src;
        unsigned short* dst;
        if (!region) { src = (j < E8) ? (w1 + j) : (sw1 + (j - E8)); dst = w1bf + j; }
        else         { src = (j < E8) ? (w2 + j) : (sw2 + (j - E8)); dst = w2bf + j; }
        float4 v = *(const float4*)src;
        v4bf b;
        b[0] = (__bf16)v.x; b[1] = (__bf16)v.y; b[2] = (__bf16)v.z; b[3] = (__bf16)v.w;
        *(v4bf*)dst = b;
    }
}

// ---- gate: scores -> softmax -> top2 -> per-expert token lists --------------
__global__ __launch_bounds__(256) void k_gate(const float* __restrict__ x,
                                              const float* __restrict__ gw,
                                              int* __restrict__ counts,
                                              int* __restrict__ tokens,
                                              float* __restrict__ wlist) {
    const int t = blockIdx.x * 8 + (threadIdx.x >> 5);   // one wave per token
    const int lane = threadIdx.x & 31;
    float acc[EE];
#pragma unroll
    for (int e = 0; e < EE; ++e) acc[e] = 0.f;
    const float* xr = x + (size_t)t * DD;
    for (int d = lane; d < DD; d += 32) {
        float xv = xr[d];
#pragma unroll
        for (int e = 0; e < EE; ++e) acc[e] += xv * gw[e * DD + d];
    }
#pragma unroll
    for (int e = 0; e < EE; ++e) {
        float v = acc[e];
        v += __shfl_xor(v, 16, 32);
        v += __shfl_xor(v,  8, 32);
        v += __shfl_xor(v,  4, 32);
        v += __shfl_xor(v,  2, 32);
        v += __shfl_xor(v,  1, 32);
        acc[e] = v;
    }
    if (lane == 0) {
        float m = acc[0];
#pragma unroll
        for (int e = 1; e < EE; ++e) m = fmaxf(m, acc[e]);
        float p[EE]; float s = 0.f;
#pragma unroll
        for (int e = 0; e < EE; ++e) { p[e] = __expf(acc[e] - m); s += p[e]; }
        float inv = 1.0f / s;
        int i1 = 0; float p1 = p[0];
#pragma unroll
        for (int e = 1; e < EE; ++e) if (p[e] > p1) { p1 = p[e]; i1 = e; }
        int i2 = (i1 == 0) ? 1 : 0; float p2 = p[i2];
#pragma unroll
        for (int e = 0; e < EE; ++e)
            if (e != i1 && p[e] > p2) { p2 = p[e]; i2 = e; }
        int pos = atomicAdd(&counts[i1], 1);
        tokens[i1 * TT + pos] = t; wlist[i1 * TT + pos] = p1 * inv;
        pos = atomicAdd(&counts[i2], 1);
        tokens[i2 * TT + pos] = t; wlist[i2 * TT + pos] = p2 * inv;
    }
}

// ---- fused grouped FFN: out += w * (gelu(x@W1^T+b1) @ W2^T + b2) ------------
// grid = (256 tiles, 9 buckets) x 256 threads (8 waves).
// Block owns 32 tokens x full D=1024 output; wave w owns 128 output columns.
// Each B fragment feeds 2 WMMAs (two 16-row M tiles) -> 2x L2-byte reuse.
__global__ __launch_bounds__(256) void k_ffn(const unsigned short* __restrict__ w1u,
                                             const unsigned short* __restrict__ w2u,
                                             const float* __restrict__ b1,
                                             const float* __restrict__ b2,
                                             const float* __restrict__ sb1,
                                             const float* __restrict__ sb2,
                                             const float* __restrict__ x,
                                             const int* __restrict__ counts,
                                             const int* __restrict__ tokens,
                                             const float* __restrict__ wlist,
                                             float* __restrict__ out) {
    const int e    = blockIdx.y;              // bucket 0..8 (8 = shared expert)
    const int tile = blockIdx.x;              // 32-token tile within bucket
    const int Ne   = (e < EE) ? counts[e] : TT;
    if (tile * MT >= Ne) return;

    __shared__ __bf16 xs[MT][1032];           // x tile, bf16 (66 KB)
    __shared__ __bf16 hb[MT][136];            // gelu hidden chunk [32 x 128]
    __shared__ int    s_tok[MT];
    __shared__ float  s_w[MT];

    const int tid  = threadIdx.x;
    const int lane = tid & 31;
    const int wv   = tid >> 5;                // wave 0..7
    const int half = lane >> 4;               // 0: lanes 0-15, 1: lanes 16-31
    const int l16  = lane & 15;

    if (tid < MT) {
        int m = tile * MT + tid;
        int tok; float w;
        if (e < EE) {
            int mm = (m < Ne) ? m : (Ne - 1);
            tok = tokens[e * TT + mm];
            w   = (m < Ne) ? wlist[e * TT + mm] : 0.0f;
        } else { tok = m; w = 1.0f; }
        s_tok[tid] = tok; s_w[tid] = w;
    }
    __syncthreads();

    // stage x tile (gather rows via token indices), convert to bf16 in LDS
    for (int i = tid; i < MT * 256; i += 256) {
        int r  = i >> 8;
        int c4 = (i & 255) << 2;
        float4 v = *(const float4*)(x + (size_t)s_tok[r] * DD + c4);
        v4bf b;
        b[0] = (__bf16)v.x; b[1] = (__bf16)v.y; b[2] = (__bf16)v.z; b[3] = (__bf16)v.w;
        *(v4bf*)&xs[r][c4] = b;
    }
    __syncthreads();

    const __bf16* W1e = reinterpret_cast<const __bf16*>(w1u) + (size_t)e * FF * DD;
    const __bf16* W2e = reinterpret_cast<const __bf16*>(w2u) + (size_t)e * DD * FF;
    const float*  b1e = (e < EE) ? (b1 + (size_t)e * FF) : sb1;
    const float*  b2e = (e < EE) ? (b2 + (size_t)e * DD) : sb2;

    const int dbase = wv * 128;               // this wave's output column base
    v8f acc0[8], acc1[8];
#pragma unroll
    for (int nt = 0; nt < 8; ++nt) {
        acc0[nt] = (v8f){0.f, 0.f, 0.f, 0.f, 0.f, 0.f, 0.f, 0.f};
        acc1[nt] = (v8f){0.f, 0.f, 0.f, 0.f, 0.f, 0.f, 0.f, 0.f};
    }

    float b2v[8];
#pragma unroll
    for (int nt = 0; nt < 8; ++nt) b2v[nt] = b2e[dbase + nt * 16 + l16];

    for (int fc = 0; fc < FF; fc += 128) {
        const int fb = fc + wv * 16;          // wave's 16 hidden columns

        // ---- GEMM1: h[32x16] = x_tile[32x1024] @ W1[fb..fb+15, :]^T --------
        v8f hc0 = (v8f){0.f, 0.f, 0.f, 0.f, 0.f, 0.f, 0.f, 0.f};
        v8f hc1 = (v8f){0.f, 0.f, 0.f, 0.f, 0.f, 0.f, 0.f, 0.f};
        const __bf16* w1row = W1e + (size_t)(fb + l16) * DD + half * 16;
        for (int k0 = 0; k0 < DD; k0 += 32) {
            v16bf b = *(const v16bf*)(w1row + k0);
            __builtin_prefetch(w1row + k0 + 128, 0, 1);
            v8bf alo = *(const v8bf*)&xs[l16][k0 + half * 16];
            v8bf ahi = *(const v8bf*)&xs[l16][k0 + half * 16 + 8];
            v16bf a0 = __builtin_shufflevector(alo, ahi,
                0, 1, 2, 3, 4, 5, 6, 7, 8, 9, 10, 11, 12, 13, 14, 15);
            alo = *(const v8bf*)&xs[16 + l16][k0 + half * 16];
            ahi = *(const v8bf*)&xs[16 + l16][k0 + half * 16 + 8];
            v16bf a1 = __builtin_shufflevector(alo, ahi,
                0, 1, 2, 3, 4, 5, 6, 7, 8, 9, 10, 11, 12, 13, 14, 15);
            hc0 = __builtin_amdgcn_wmma_f32_16x16x32_bf16(
                false, a0, false, b, (short)0, hc0, false, false);
            hc1 = __builtin_amdgcn_wmma_f32_16x16x32_bf16(
                false, a1, false, b, (short)0, hc1, false, false);
        }

        // ---- bias + exact GELU, write hidden chunk to LDS as bf16 ----------
        const float bias1 = b1e[fb + l16];
        __syncthreads();   // previous GEMM2 finished reading hb
#pragma unroll
        for (int r = 0; r < 8; ++r) {
            float v0 = hc0[r] + bias1;
            float v1 = hc1[r] + bias1;
            float g0 = 0.5f * v0 * (1.0f + erff(v0 * 0.70710678118654752f));
            float g1 = 0.5f * v1 * (1.0f + erff(v1 * 0.70710678118654752f));
            hb[r + 8 * half][wv * 16 + l16]      = (__bf16)g0;
            hb[16 + r + 8 * half][wv * 16 + l16] = (__bf16)g1;
        }
        __syncthreads();

        // ---- GEMM2: acc[32x128] += h[32x128] @ W2[dbase..+127, fc..+127]^T -
#pragma unroll
        for (int ks = 0; ks < 4; ++ks) {
            const int kk = ks * 32;
            v8bf alo = *(const v8bf*)&hb[l16][kk + half * 16];
            v8bf ahi = *(const v8bf*)&hb[l16][kk + half * 16 + 8];
            v16bf a20 = __builtin_shufflevector(alo, ahi,
                0, 1, 2, 3, 4, 5, 6, 7, 8, 9, 10, 11, 12, 13, 14, 15);
            alo = *(const v8bf*)&hb[16 + l16][kk + half * 16];
            ahi = *(const v8bf*)&hb[16 + l16][kk + half * 16 + 8];
            v16bf a21 = __builtin_shufflevector(alo, ahi,
                0, 1, 2, 3, 4, 5, 6, 7, 8, 9, 10, 11, 12, 13, 14, 15);
            const __bf16* w2p = W2e + (size_t)fc + kk + half * 16;
#pragma unroll
            for (int nt = 0; nt < 8; ++nt) {
                const int d = dbase + nt * 16 + l16;
                v16bf b = *(const v16bf*)(w2p + (size_t)d * FF);
                acc0[nt] = __builtin_amdgcn_wmma_f32_16x16x32_bf16(
                    false, a20, false, b, (short)0, acc0[nt], false, false);
                acc1[nt] = __builtin_amdgcn_wmma_f32_16x16x32_bf16(
                    false, a21, false, b, (short)0, acc1[nt], false, false);
            }
        }
    }

    // ---- epilogue: scatter-add w * (acc + b2) into out ----------------------
#pragma unroll
    for (int r = 0; r < 8; ++r) {
        const int row0 = r + 8 * half;
        const int row1 = 16 + row0;
        const int tok0 = s_tok[row0];
        const int tok1 = s_tok[row1];
        const float w0 = s_w[row0];
        const float w1w = s_w[row1];
        float* op0 = out + (size_t)tok0 * DD;
        float* op1 = out + (size_t)tok1 * DD;
#pragma unroll
        for (int nt = 0; nt < 8; ++nt) {
            const int d = dbase + nt * 16 + l16;
            unsafeAtomicAdd(op0 + d, w0  * (acc0[nt][r] + b2v[nt]));
            unsafeAtomicAdd(op1 + d, w1w * (acc1[nt][r] + b2v[nt]));
        }
    }
}

// ---- host launcher ----------------------------------------------------------
extern "C" void kernel_launch(void* const* d_in, const int* in_sizes, int n_in,
                              void* d_out, int out_size, void* d_ws, size_t ws_size,
                              hipStream_t stream) {
    const float* x      = (const float*)d_in[0];
    const float* gate_w = (const float*)d_in[1];
    const float* w1     = (const float*)d_in[2];
    const float* b1     = (const float*)d_in[3];
    const float* w2     = (const float*)d_in[4];
    const float* b2     = (const float*)d_in[5];
    const float* sw1    = (const float*)d_in[6];
    const float* sb1    = (const float*)d_in[7];
    const float* sw2    = (const float*)d_in[8];
    const float* sb2    = (const float*)d_in[9];
    float* out = (float*)d_out;

    char* ws = (char*)d_ws;
    unsigned short* w1bf = (unsigned short*)ws;
    unsigned short* w2bf = (unsigned short*)(ws + OFF_W2);
    int*   counts = (int*)(ws + OFF_CNT);
    int*   tokens = (int*)(ws + OFF_TOK);
    float* wlist  = (float*)(ws + OFF_WL);

    // zero output accumulator and expert counters
    k_zero<<<2048, 256, 0, stream>>>((float4*)out, TT * DD / 4);
    k_zero<<<1, 32, 0, stream>>>((float4*)counts, 2);   // 8 ints = 2 float4

    // bf16 weight conversion (151 MB -> L2 resident)
    k_convert<<<4096, 256, 0, stream>>>(w1, sw1, w2, sw2, w1bf, w2bf);

    // gating + routing lists
    k_gate<<<TT / 8, 256, 0, stream>>>(x, gate_w, counts, tokens, wlist);

    // grouped fused FFN over 9 buckets (8 routed + shared)
    dim3 grid(TT / MT, NB);
    k_ffn<<<grid, 256, 0, stream>>>(w1bf, w2bf, b1, b2, sb1, sb2,
                                    x, counts, tokens, wlist, out);
}
